// Transformer_postnorm_cross_spd_63299228008808
// MI455X (gfx1250) — matrix operand order repaired and checked by reading, not compiled, not run on hardware
//
#include <hip/hip_runtime.h>
#include <hip/hip_bf16.h>

typedef __attribute__((ext_vector_type(16))) __bf16 v16bf;
typedef __attribute__((ext_vector_type(8)))  __bf16 v8bf;
typedef __attribute__((ext_vector_type(8)))  float  v8f;
typedef unsigned int u32x4 __attribute__((ext_vector_type(4)));
typedef int          i32x8 __attribute__((ext_vector_type(8)));
typedef int          i32x4 __attribute__((ext_vector_type(4)));

#define DEV __device__ __forceinline__

static constexpr int Bq = 8, Nseq = 1024, Dm = 512, Hh = 8, Mlp = 2048, Depth = 6;
static constexpr int Mrows = Bq * Nseq; // 8192

#if __has_builtin(__builtin_amdgcn_tensor_load_to_lds)
#define USE_TDM 1
#else
#define USE_TDM 0
#endif

// ---------------------------------------------------------------------------
// TDM: DMA a 2-D bf16 tile [tile_d1 rows x tile_d0 elems] from global memory
// (row pitch d0_stride elems) into LDS with a 16B pad every 128B -> LDS row
// pitch of 72 bf16 elements, matching the WMMA fragment readers below.
// Descriptor layout per CDNA5 ISA ch.8 (D# groups 0/1; groups 2/3 zero = 2D).
// Issued by one wave; EXEC is ignored for tensor ops; completion via
// s_wait_tensorcnt + workgroup barrier.
// 6-arg builtin variant (clang-23 / therock-10.0 headers).
// ---------------------------------------------------------------------------
#if USE_TDM
DEV void tdm_load_tile_bf16(const __bf16* gsrc, void* lds_dst,
                            unsigned tile_d0, unsigned tile_d1,
                            unsigned d0_stride_elems) {
  unsigned long long ga = (unsigned long long)(uintptr_t)gsrc;
  unsigned lds_addr = (unsigned)(uintptr_t)lds_dst;  // low 32 bits = LDS offset
  u32x4 g0;
  g0[0] = 1u;                                   // count=1, user D#, no gather
  g0[1] = lds_addr;                             // lds_addr (bytes)
  g0[2] = (unsigned)(ga & 0xffffffffu);         // global_addr[31:0]
  g0[3] = (unsigned)((ga >> 32) & 0x01ffffffu)  // global_addr[56:32]
          | (2u << 30);                         // type=2 ("image")
  i32x8 g1;
  g1[0] = (int)((1u << 16)     // data_size=1 -> 2 bytes
                | (1u << 20)   // pad_enable
                | (4u << 22)   // pad_interval: 2^(4+1)=32 DWORDs = 128B
                | (3u << 25)); // pad_amount: 4 DWORDs = 16B
  g1[1] = (int)((tile_d0 & 0xffffu) << 16);                    // tensor_dim0 lo
  g1[2] = (int)(((tile_d0 >> 16) & 0xffffu) |
                ((tile_d1 & 0xffffu) << 16));                  // dim0 hi | dim1 lo
  g1[3] = (int)(((tile_d1 >> 16) & 0xffffu) |
                ((tile_d0 & 0xffffu) << 16));                  // dim1 hi | tile_dim0
  g1[4] = (int)(tile_d1 & 0xffffu);                            // tile_dim1 (tile_dim2=0)
  g1[5] = (int)d0_stride_elems;                                // tensor_dim0_stride[31:0]
  g1[6] = 0;                                                   // stride[47:32] | dim1_stride lo
  g1[7] = 0;
  i32x4 z4 = {0, 0, 0, 0};
  i32x8 z8 = {0, 0, 0, 0, 0, 0, 0, 0};
  __builtin_amdgcn_tensor_load_to_lds(g0, g1, z4, z4, z8, 0);
}
#endif

// ---------------------------------------------------------------------------
// WMMA fragment helpers (CDNA5 16x16x32 bf16 layouts, ISA 7.12.2)
// ---------------------------------------------------------------------------
DEV v16bf frag_a(const __bf16* base, int ld, int m0, int ks, int lane) {
  const __bf16* p = base + (m0 + (lane & 15)) * ld + ks + ((lane & 16) ? 8 : 0);
  v8bf lo = *(const v8bf*)(p);
  v8bf hi = *(const v8bf*)(p + 16);
  v16bf r;
#pragma unroll
  for (int i = 0; i < 8; ++i) { r[i] = lo[i]; r[i + 8] = hi[i]; }
  return r;
}

DEV v16bf frag_b(const __bf16* base, int ld, int n0, int ks, int lane) {
  const __bf16* p = base + (n0 + (lane & 15)) * ld + ks + ((lane & 16) ? 8 : 0);
  v8bf lo = *(const v8bf*)(p);
  v8bf hi = *(const v8bf*)(p + 16);
  v16bf r;
#pragma unroll
  for (int i = 0; i < 8; ++i) { r[i] = lo[i]; r[i + 8] = hi[i]; }
  return r;
}

DEV v8f wmma_bf16(v16bf a, v16bf b, v8f c) {
  return __builtin_amdgcn_wmma_f32_16x16x32_bf16(false, a, false, b, (short)0, c,
                                                 false, false);
}

// ---------------------------------------------------------------------------
// GEMM: C[M,Nc] = A[M,K](bf16) * Bt[Nc,K](bf16 pre-transposed), f32 accum.
// 128x128x64 block tile, 8 waves, wave tile 32x64. Tiles staged by TDM.
// ---------------------------------------------------------------------------
template <bool BIAS, bool GELU_, bool RES, bool OUTF, bool OUTB>
__global__ __launch_bounds__(256) void gemm_bf16_kernel(
    const __bf16* __restrict__ A, const __bf16* __restrict__ Bt,
    const float* __restrict__ bias, const float* __restrict__ res,
    float* __restrict__ outf, __bf16* __restrict__ outb, int M, int K, int Nc) {
  __shared__ alignas(16) __bf16 As[128 * 72];
  __shared__ alignas(16) __bf16 Bs[128 * 72];
  const int tid = threadIdx.x;
  const int lane = tid & 31;
  const int wid = tid >> 5;
  const int wm = (wid & 3) * 32;
  const int wn = (wid >> 2) * 64;
  const int m0 = blockIdx.y * 128;
  const int n0 = blockIdx.x * 128;

  v8f acc[2][4];
#pragma unroll
  for (int i = 0; i < 2; ++i)
#pragma unroll
    for (int j = 0; j < 4; ++j) acc[i][j] = v8f{};

  for (int k0 = 0; k0 < K; k0 += 64) {
    __syncthreads();
#if USE_TDM
    if (wid == 0) {
      tdm_load_tile_bf16(&A[(size_t)m0 * K + k0], As, 64, 128, (unsigned)K);
      __builtin_amdgcn_s_wait_tensorcnt(0);
    } else if (wid == 1) {
      tdm_load_tile_bf16(&Bt[(size_t)n0 * K + k0], Bs, 64, 128, (unsigned)K);
      __builtin_amdgcn_s_wait_tensorcnt(0);
    }
#else
#pragma unroll
    for (int c = tid; c < 1024; c += 256) {
      int r = c >> 3, c8 = (c & 7) * 8;
      *(v8bf*)(&As[r * 72 + c8]) =
          *(const v8bf*)(&A[(size_t)(m0 + r) * K + k0 + c8]);
      *(v8bf*)(&Bs[r * 72 + c8]) =
          *(const v8bf*)(&Bt[(size_t)(n0 + r) * K + k0 + c8]);
    }
#endif
    __syncthreads();
#pragma unroll
    for (int ks = 0; ks < 64; ks += 32) {
      v16bf a0 = frag_a(As, 72, wm, ks, lane);
      v16bf a1 = frag_a(As, 72, wm + 16, ks, lane);
#pragma unroll
      for (int nf = 0; nf < 4; ++nf) {
        v16bf bfr = frag_b(Bs, 72, wn + nf * 16, ks, lane);
        acc[0][nf] = wmma_bf16(a0, bfr, acc[0][nf]);
        acc[1][nf] = wmma_bf16(a1, bfr, acc[1][nf]);
      }
    }
  }

  const int rowoff = (lane & 16) ? 8 : 0;
#pragma unroll
  for (int mf = 0; mf < 2; ++mf)
#pragma unroll
    for (int nf = 0; nf < 4; ++nf)
#pragma unroll
      for (int v = 0; v < 8; ++v) {
        int row = m0 + wm + mf * 16 + v + rowoff;
        int col = n0 + wn + nf * 16 + (lane & 15);
        float x = acc[mf][nf][v];
        if (BIAS) x += bias[col];
        if (GELU_) x = 0.5f * x * (1.f + erff(x * 0.70710678118654752440f));
        size_t o = (size_t)row * Nc + col;
        if (RES) x += res[o];
        if (OUTF) outf[o] = x;
        if (OUTB) outb[o] = (__bf16)x;
      }
}

// ---------------------------------------------------------------------------
// Flash-style attention, one (b,h) per blockIdx.{z,y}, 128 queries/block,
// 16 queries/wave, keytile=64, online softmax, spd bias added in f32.
// K tile staged via TDM (overlapped with V transpose scatter).
// ---------------------------------------------------------------------------
__global__ __launch_bounds__(256) void attn_kernel(
    const __bf16* __restrict__ Qb, int qstride, const __bf16* __restrict__ Kb,
    int kstride, const __bf16* __restrict__ Vb, int vstride,
    const float* __restrict__ spd, __bf16* __restrict__ Ob, int ostride) {
  __shared__ alignas(16) __bf16 Kt[64 * 72];
  __shared__ alignas(16) __bf16 Vt[64 * 72];  // transposed: [dh][key]
  __shared__ alignas(16) __bf16 Ps[8][16 * 72];
  const int tid = threadIdx.x, lane = tid & 31, wid = tid >> 5;
  const int b = blockIdx.z, h = blockIdx.y;
  const int q0 = blockIdx.x * 128 + wid * 16;
  const int hc = h * 64;
  const int rowoff = (lane & 16) ? 8 : 0;

  // Q fragments (2 k-steps of 32), pre-scaled by DH^-0.5 = 2^-3 (exact in bf16)
  v16bf qf[2];
  {
    int r = q0 + (lane & 15);
    const __bf16* qrow = Qb + (size_t)(b * Nseq + r) * qstride + hc;
#pragma unroll
    for (int ksi = 0; ksi < 2; ++ksi) {
      int kb = ksi * 32 + ((lane & 16) ? 8 : 0);
      v8bf lo = *(const v8bf*)(qrow + kb);
      v8bf hi = *(const v8bf*)(qrow + kb + 16);
#pragma unroll
      for (int i = 0; i < 8; ++i) {
        qf[ksi][i] = (__bf16)((float)lo[i] * 0.125f);
        qf[ksi][i + 8] = (__bf16)((float)hi[i] * 0.125f);
      }
    }
  }

  float mrow[8], lrow[8];
  v8f o[4];
#pragma unroll
  for (int v = 0; v < 8; ++v) { mrow[v] = -1e30f; lrow[v] = 0.f; }
#pragma unroll
  for (int nf = 0; nf < 4; ++nf) o[nf] = v8f{};

  for (int j0 = 0; j0 < Nseq; j0 += 64) {
    __syncthreads();
#if USE_TDM
    if (wid == 0)
      tdm_load_tile_bf16(&Kb[(size_t)(b * Nseq + j0) * kstride + hc], Kt, 64,
                         64, (unsigned)kstride);
#endif
    for (int c = tid; c < 512; c += 256) {
      int r = c >> 3, c8 = (c & 7) * 8;
#if !USE_TDM
      *(v8bf*)(&Kt[r * 72 + c8]) =
          *(const v8bf*)(&Kb[(size_t)(b * Nseq + j0 + r) * kstride + hc + c8]);
#endif
      v8bf vv = *(const v8bf*)(&Vb[(size_t)(b * Nseq + j0 + r) * vstride + hc + c8]);
#pragma unroll
      for (int i = 0; i < 8; ++i) Vt[(c8 + i) * 72 + r] = vv[i];
    }
#if USE_TDM
    if (wid == 0) __builtin_amdgcn_s_wait_tensorcnt(0);
#endif
    __syncthreads();

    v8f s[4];
#pragma unroll
    for (int nf = 0; nf < 4; ++nf) s[nf] = v8f{};
#pragma unroll
    for (int ksi = 0; ksi < 2; ++ksi)
#pragma unroll
      for (int nf = 0; nf < 4; ++nf)
        s[nf] = wmma_bf16(qf[ksi], frag_b(Kt, 72, nf * 16, ksi * 32, lane), s[nf]);

    // add spd bias, online softmax
    float p[4][8], mloc[8];
#pragma unroll
    for (int v = 0; v < 8; ++v) mloc[v] = -1e30f;
#pragma unroll
    for (int nf = 0; nf < 4; ++nf)
#pragma unroll
      for (int v = 0; v < 8; ++v) {
        int qi = q0 + v + rowoff;
        int j = j0 + nf * 16 + (lane & 15);
        float x = s[nf][v] + spd[(size_t)(b * Nseq + qi) * Nseq + j];
        p[nf][v] = x;
        mloc[v] = fmaxf(mloc[v], x);
      }
#pragma unroll
    for (int v = 0; v < 8; ++v) {  // a row lives in one 16-lane half
      mloc[v] = fmaxf(mloc[v], __shfl_xor(mloc[v], 1, 32));
      mloc[v] = fmaxf(mloc[v], __shfl_xor(mloc[v], 2, 32));
      mloc[v] = fmaxf(mloc[v], __shfl_xor(mloc[v], 4, 32));
      mloc[v] = fmaxf(mloc[v], __shfl_xor(mloc[v], 8, 32));
    }
    float corr[8], lsum[8];
#pragma unroll
    for (int v = 0; v < 8; ++v) {
      float mn = fmaxf(mrow[v], mloc[v]);
      corr[v] = __expf(mrow[v] - mn);
      mrow[v] = mn;
      lsum[v] = 0.f;
    }
#pragma unroll
    for (int nf = 0; nf < 4; ++nf)
#pragma unroll
      for (int v = 0; v < 8; ++v) {
        float e = __expf(p[nf][v] - mrow[v]);
        p[nf][v] = e;
        lsum[v] += e;
      }
#pragma unroll
    for (int v = 0; v < 8; ++v) {
      lsum[v] += __shfl_xor(lsum[v], 1, 32);
      lsum[v] += __shfl_xor(lsum[v], 2, 32);
      lsum[v] += __shfl_xor(lsum[v], 4, 32);
      lsum[v] += __shfl_xor(lsum[v], 8, 32);
      lrow[v] = lrow[v] * corr[v] + lsum[v];
    }
#pragma unroll
    for (int nf = 0; nf < 4; ++nf)
#pragma unroll
      for (int v = 0; v < 8; ++v) o[nf][v] *= corr[v];

    // C-layout -> A-layout via per-wave private LDS strip (same-wave DS in order)
    __bf16* Pw = &Ps[wid][0];
#pragma unroll
    for (int nf = 0; nf < 4; ++nf)
#pragma unroll
      for (int v = 0; v < 8; ++v)
        Pw[(v + rowoff) * 72 + nf * 16 + (lane & 15)] = (__bf16)p[nf][v];
#pragma unroll
    for (int ksi = 0; ksi < 2; ++ksi) {
      v16bf pa = frag_a(Pw, 72, 0, ksi * 32, lane);
#pragma unroll
      for (int nf = 0; nf < 4; ++nf)
        o[nf] = wmma_bf16(pa, frag_b(Vt, 72, nf * 16, ksi * 32, lane), o[nf]);
    }
  }

#pragma unroll
  for (int nf = 0; nf < 4; ++nf)
#pragma unroll
    for (int v = 0; v < 8; ++v) {
      int qi = q0 + v + rowoff;
      float val = o[nf][v] / lrow[v];
      Ob[(size_t)(b * Nseq + qi) * ostride + hc + nf * 16 + (lane & 15)] =
          (__bf16)val;
    }
}

// ---------------------------------------------------------------------------
// LayerNorm over D=512, one row per block, writes f32 state + bf16 mirror.
// ---------------------------------------------------------------------------
__global__ __launch_bounds__(256) void layernorm_kernel(
    const float* __restrict__ in, const float* __restrict__ g,
    const float* __restrict__ bb, float* __restrict__ outf,
    __bf16* __restrict__ outb) {
  __shared__ float s1[256], s2[256];
  const int row = blockIdx.x, tid = threadIdx.x;
  const float* r = in + (size_t)row * Dm;
  float v0 = r[tid], v1 = r[tid + 256];
  s1[tid] = v0 + v1;
  s2[tid] = v0 * v0 + v1 * v1;
  __syncthreads();
  for (int s = 128; s > 0; s >>= 1) {
    if (tid < s) { s1[tid] += s1[tid + s]; s2[tid] += s2[tid + s]; }
    __syncthreads();
  }
  float mean = s1[0] * (1.f / Dm);
  float var = s2[0] * (1.f / Dm) - mean * mean;
  float rstd = rsqrtf(var + 1e-5f);
  float o0 = (v0 - mean) * rstd * g[tid] + bb[tid];
  float o1 = (v1 - mean) * rstd * g[tid + 256] + bb[tid + 256];
  float* of = outf + (size_t)row * Dm;
  __bf16* ob = outb + (size_t)row * Dm;
  of[tid] = o0;
  of[tid + 256] = o1;
  ob[tid] = (__bf16)o0;
  ob[tid + 256] = (__bf16)o1;
}

// Weight [K][Nc] f32 -> [Nc][K] bf16, 16x16 tiles, no div/mod
__global__ void transpose_cvt_kernel(const float* __restrict__ in,
                                     __bf16* __restrict__ out, int K, int Nc) {
  int n = blockIdx.x * 16 + threadIdx.x;
  int k = blockIdx.y * 16 + threadIdx.y;
  out[(size_t)n * K + k] = (__bf16)in[(size_t)k * Nc + n];
}

__global__ void cvt_kernel(const float* __restrict__ in,
                           __bf16* __restrict__ out, size_t n) {
  size_t i = (size_t)blockIdx.x * blockDim.x + threadIdx.x;
  if (i < n) out[i] = (__bf16)in[i];
}

// ---------------------------------------------------------------------------
extern "C" void kernel_launch(void* const* d_in, const int* in_sizes, int n_in,
                              void* d_out, int out_size, void* d_ws,
                              size_t ws_size, hipStream_t stream) {
  (void)in_sizes; (void)n_in; (void)out_size; (void)ws_size;
  const float* x    = (const float*)d_in[0];
  const float* spd  = (const float*)d_in[1];
  const float* x2   = (const float*)d_in[2];
  const float* Wqkv = (const float*)d_in[4];
  const float* Wo   = (const float*)d_in[5];
  const float* bo   = (const float*)d_in[6];
  const float* W1   = (const float*)d_in[7];
  const float* b1   = (const float*)d_in[8];
  const float* W2   = (const float*)d_in[9];
  const float* b2   = (const float*)d_in[10];
  const float* cWq  = (const float*)d_in[11];
  const float* cWkv = (const float*)d_in[12];
  const float* cWo  = (const float*)d_in[13];
  const float* cbo  = (const float*)d_in[14];
  const float* lng  = (const float*)d_in[15];
  const float* lnb  = (const float*)d_in[16];

  char* pw = (char*)d_ws;
  auto alloc = [&](size_t bytes) -> void* {
    void* r = (void*)pw;
    pw += (bytes + 255) & ~(size_t)255;
    return r;
  };
  float*  xw    = (float*)alloc((size_t)Mrows * Dm * 4);
  float*  pre   = (float*)alloc((size_t)Mrows * Dm * 4);
  __bf16* xb    = (__bf16*)alloc((size_t)Mrows * Dm * 2);
  __bf16* x2b   = (__bf16*)alloc((size_t)Mrows * Dm * 2);
  __bf16* qkvb  = (__bf16*)alloc((size_t)Mrows * 3 * Dm * 2);
  __bf16* attnb = (__bf16*)alloc((size_t)Mrows * Dm * 2);
  __bf16* hb    = (__bf16*)alloc((size_t)Mrows * Mlp * 2);
  __bf16* cqb   = (__bf16*)alloc((size_t)Mrows * Dm * 2);
  __bf16* ckvb  = (__bf16*)alloc((size_t)Mrows * 2 * Dm * 2);
  __bf16* WqkvT = (__bf16*)alloc((size_t)Depth * 3 * Dm * Dm * 2);
  __bf16* WoT   = (__bf16*)alloc((size_t)Depth * Dm * Dm * 2);
  __bf16* W1T   = (__bf16*)alloc((size_t)Depth * Mlp * Dm * 2);
  __bf16* W2T   = (__bf16*)alloc((size_t)Depth * Dm * Mlp * 2);
  __bf16* cWqT  = (__bf16*)alloc((size_t)Dm * Dm * 2);
  __bf16* cWkvT = (__bf16*)alloc((size_t)2 * Dm * Dm * 2);
  __bf16* cWoT  = (__bf16*)alloc((size_t)Dm * Dm * 2);

  auto tcvt = [&](const float* in, __bf16* out, int K, int Nc) {
    transpose_cvt_kernel<<<dim3(Nc / 16, K / 16), dim3(16, 16), 0, stream>>>(
        in, out, K, Nc);
  };
  for (int l = 0; l < Depth; ++l) {
    tcvt(Wqkv + (size_t)l * Dm * 3 * Dm, WqkvT + (size_t)l * 3 * Dm * Dm, Dm, 3 * Dm);
    tcvt(Wo + (size_t)l * Dm * Dm, WoT + (size_t)l * Dm * Dm, Dm, Dm);
    tcvt(W1 + (size_t)l * Dm * Mlp, W1T + (size_t)l * Mlp * Dm, Dm, Mlp);
    tcvt(W2 + (size_t)l * Mlp * Dm, W2T + (size_t)l * Dm * Mlp, Mlp, Dm);
  }
  tcvt(cWq, cWqT, Dm, Dm);
  tcvt(cWkv, cWkvT, Dm, 2 * Dm);
  tcvt(cWo, cWoT, Dm, Dm);

  const size_t nact = (size_t)Mrows * Dm;
  cvt_kernel<<<dim3((unsigned)((nact + 255) / 256)), 256, 0, stream>>>(x, xb, nact);
  cvt_kernel<<<dim3((unsigned)((nact + 255) / 256)), 256, 0, stream>>>(x2, x2b, nact);
  (void)hipMemcpyAsync(xw, x, nact * 4, hipMemcpyDeviceToDevice, stream);

  const dim3 blk(256);
  // cross K/V hoisted: ckv = x2 @ cWkv
  gemm_bf16_kernel<false, false, false, false, true>
      <<<dim3(2 * Dm / 128, Mrows / 128), blk, 0, stream>>>(
          x2b, cWkvT, nullptr, nullptr, nullptr, ckvb, Mrows, Dm, 2 * Dm);

  for (int l = 0; l < Depth; ++l) {
    // qkv = x @ Wqkv[l]
    gemm_bf16_kernel<false, false, false, false, true>
        <<<dim3(3 * Dm / 128, Mrows / 128), blk, 0, stream>>>(
            xb, WqkvT + (size_t)l * 3 * Dm * Dm, nullptr, nullptr, nullptr,
            qkvb, Mrows, Dm, 3 * Dm);
    // self-attention with spd
    attn_kernel<<<dim3(Nseq / 128, Hh, Bq), blk, 0, stream>>>(
        qkvb, 3 * Dm, qkvb + Dm, 3 * Dm, qkvb + 2 * Dm, 3 * Dm, spd, attnb, Dm);
    // out proj + bo + residual
    gemm_bf16_kernel<true, false, true, true, false>
        <<<dim3(Dm / 128, Mrows / 128), blk, 0, stream>>>(
            attnb, WoT + (size_t)l * Dm * Dm, bo + (size_t)l * Dm, xw, pre,
            nullptr, Mrows, Dm, Dm);
    layernorm_kernel<<<Mrows, blk, 0, stream>>>(pre, lng, lnb, xw, xb);
    // FFN
    gemm_bf16_kernel<true, true, false, false, true>
        <<<dim3(Mlp / 128, Mrows / 128), blk, 0, stream>>>(
            xb, W1T + (size_t)l * Mlp * Dm, b1 + (size_t)l * Mlp, nullptr,
            nullptr, hb, Mrows, Dm, Mlp);
    gemm_bf16_kernel<true, false, true, true, false>
        <<<dim3(Dm / 128, Mrows / 128), blk, 0, stream>>>(
            hb, W2T + (size_t)l * Dm * Mlp, b2 + (size_t)l * Dm, xw, pre,
            nullptr, Mrows, Mlp, Dm);
    layernorm_kernel<<<Mrows, blk, 0, stream>>>(pre, lng, lnb, xw, xb);
    // cross attention (shared params)
    gemm_bf16_kernel<false, false, false, false, true>
        <<<dim3(Dm / 128, Mrows / 128), blk, 0, stream>>>(
            xb, cWqT, nullptr, nullptr, nullptr, cqb, Mrows, Dm, Dm);
    attn_kernel<<<dim3(Nseq / 128, Hh, Bq), blk, 0, stream>>>(
        cqb, Dm, ckvb, 2 * Dm, ckvb + Dm, 2 * Dm, spd, attnb, Dm);
    gemm_bf16_kernel<true, false, true, true, false>
        <<<dim3(Dm / 128, Mrows / 128), blk, 0, stream>>>(
            attnb, cWoT, cbo, xw, pre, nullptr, Mrows, Dm, Dm);
    layernorm_kernel<<<Mrows, blk, 0, stream>>>(pre, lng, lnb, xw, xb);
  }
  (void)hipMemcpyAsync(d_out, xw, nact * 4, hipMemcpyDeviceToDevice, stream);
}
